// Network_90787018702926
// MI455X (gfx1250) — compile-verified, matrix operand order checked by reading, not run on hardware
//
#include <hip/hip_runtime.h>
#include <hip/hip_bf16.h>

// ---------------------------------------------------------------------------
// Model dims (from reference)
// ---------------------------------------------------------------------------
#define NN      1365
#define TSEQ    32
#define FEAT    768
#define HID     512
#define ATTN    256
#define THID    128
#define CLIN    1024     // HID + 4*THID
#define TTIME   512

typedef __attribute__((ext_vector_type(16))) _Float16 v16h;
typedef __attribute__((ext_vector_type(8)))  _Float16 v8h;
typedef __attribute__((ext_vector_type(8)))  float    v8f;

// ---------------------------------------------------------------------------
// WMMA GEMM:  C[M,N] = act( A[M,K] * W[N,K]^T + bias )
//   A, W fp16 staging; fp32 accumulate. One wave owns a 16x64 output tile:
//   per K-step = 2 A loads + 8 B loads + 4 v_wmma (A reused 4x).
//   Tile index is COLUMN-major over (tm,tn): the 8 waves of a block share the
//   same B (weight) panel -> weight loads hit the WGP cache instead of L2.
//   Requires: K % 32 == 0, N % 64 == 0 (true for all GEMMs routed here).
//   M ragged: A row pointer clamped to M-1 (garbage rows only pollute
//   unstored C rows), stores guarded by m < M. No masking in the K loop.
//   All row strides are multiples of 8 halves -> 16B-aligned v8h loads.
// ---------------------------------------------------------------------------
__global__ void wmma_gemm(const _Float16* __restrict__ A, int lda,
                          const _Float16* __restrict__ W,
                          const float* __restrict__ bias,
                          float* __restrict__ C,          // may be null
                          _Float16* __restrict__ Ch,      // fp16 mirror, may be null
                          int ldc,
                          int M, int N, int K, int relu)
{
    const int lane   = threadIdx.x & 31;
    const int wid    = threadIdx.x >> 5;
    const int tilesM = (M + 15) >> 4;
    const int tilesN = N >> 6;                   // 64-wide N tiles
    const int tile   = blockIdx.x * (blockDim.x >> 5) + wid;
    if (tile >= tilesM * tilesN) return;         // wave-uniform: EXEC stays full

    const int tn = tile / tilesM;                // column-major: block shares B
    const int tm = tile - tn * tilesM;
    const int r  = lane & 15;
    const int hi = (lane >> 4) << 3;             // 0 (lanes 0-15) or 8 (16-31)

    int arow = tm * 16 + r;
    arow = arow < M ? arow : M - 1;              // clamp; garbage masked at store
    const int brow = tn * 64 + r;                // +16*j < N (N % 64 == 0)

    const _Float16* __restrict__ Ap  = A + (size_t)arow * lda;
    const _Float16* __restrict__ Bp0 = W + (size_t)(brow     ) * K;
    const _Float16* __restrict__ Bp1 = W + (size_t)(brow + 16) * K;
    const _Float16* __restrict__ Bp2 = W + (size_t)(brow + 32) * K;
    const _Float16* __restrict__ Bp3 = W + (size_t)(brow + 48) * K;

#define LD16(P, kk) __builtin_shufflevector(*(const v8h*)((P) + (kk) + hi),          \
                                            *(const v8h*)((P) + (kk) + 16 + hi),    \
                                            0,1,2,3,4,5,6,7,8,9,10,11,12,13,14,15)
    v8f acc0 = {}, acc1 = {}, acc2 = {}, acc3 = {};
    for (int kk = 0; kk < K; kk += 32) {
        const v16h a  = LD16(Ap, kk);
        const v16h b0 = LD16(Bp0, kk);
        const v16h b1 = LD16(Bp1, kk);
        const v16h b2 = LD16(Bp2, kk);
        const v16h b3 = LD16(Bp3, kk);
        acc0 = __builtin_amdgcn_wmma_f32_16x16x32_f16(false, a, false, b0, (short)0, acc0, false, false);
        acc1 = __builtin_amdgcn_wmma_f32_16x16x32_f16(false, a, false, b1, (short)0, acc1, false, false);
        acc2 = __builtin_amdgcn_wmma_f32_16x16x32_f16(false, a, false, b2, (short)0, acc2, false, false);
        acc3 = __builtin_amdgcn_wmma_f32_16x16x32_f16(false, a, false, b3, (short)0, acc3, false, false);
    }
#undef LD16

    const int col0 = tn * 64 + r;
    const float bv0 = bias ? bias[col0]      : 0.f;
    const float bv1 = bias ? bias[col0 + 16] : 0.f;
    const float bv2 = bias ? bias[col0 + 32] : 0.f;
    const float bv3 = bias ? bias[col0 + 48] : 0.f;
#pragma unroll
    for (int v = 0; v < 8; ++v) {                // C: M = v + 8*(lane>>4)
        const int m = tm * 16 + v + ((lane >> 4) << 3);
        if (m < M) {
            float x0 = acc0[v] + bv0;
            float x1 = acc1[v] + bv1;
            float x2 = acc2[v] + bv2;
            float x3 = acc3[v] + bv3;
            if (relu) {
                x0 = fmaxf(x0, 0.f); x1 = fmaxf(x1, 0.f);
                x2 = fmaxf(x2, 0.f); x3 = fmaxf(x3, 0.f);
            }
            const size_t base = (size_t)m * ldc + col0;
            if (C) {
                C[base] = x0; C[base + 16] = x1; C[base + 32] = x2; C[base + 48] = x3;
            }
            if (Ch) {
                Ch[base]      = (_Float16)x0; Ch[base + 16] = (_Float16)x1;
                Ch[base + 32] = (_Float16)x2; Ch[base + 48] = (_Float16)x3;
            }
        }
    }
}

// ---------------------------------------------------------------------------
// fp32 -> fp16 staging conversion
// ---------------------------------------------------------------------------
__global__ void f32_to_f16(const float* __restrict__ src, _Float16* __restrict__ dst, int n)
{
    const int i = blockIdx.x * blockDim.x + threadIdx.x;
    if (i < n) dst[i] = (_Float16)src[i];
}

__global__ void zero_f32(float* __restrict__ p, int n)
{
    const int i = blockIdx.x * blockDim.x + threadIdx.x;
    if (i < n) p[i] = 0.f;
}

__global__ void zero_f16(_Float16* __restrict__ p, int n)
{
    const int i = blockIdx.x * blockDim.x + threadIdx.x;
    if (i < n) p[i] = (_Float16)0.f;
}

// ---------------------------------------------------------------------------
// Fused GRU gate math: h = (1-z)*n + z*h_prev
//   gh may be null => 0 contribution; hprev may be null => 0.
//   Optional fp16 mirror of h for the next GEMM. In-place safe.
// ---------------------------------------------------------------------------
__global__ void gru_gate(const float* __restrict__ gi,
                         const float* __restrict__ gh,
                         const float* __restrict__ bih,
                         const float* __restrict__ bhh,
                         const float* __restrict__ hprev,
                         float* __restrict__ hout,
                         _Float16* __restrict__ hout_h,   // may be null
                         int rows, int H)
{
    const int idx = blockIdx.x * blockDim.x + threadIdx.x;
    if (idx >= rows * H) return;
    const int row = idx / H;
    const int j   = idx - row * H;
    const float* g = gi + (size_t)row * 3 * H;
    float ir = g[j]         + bih[j];
    float iz = g[H + j]     + bih[H + j];
    float in = g[2 * H + j] + bih[2 * H + j];
    float hr = bhh[j], hz = bhh[H + j], hn = bhh[2 * H + j];
    if (gh) {
        const float* q = gh + (size_t)row * 3 * H;
        hr += q[j]; hz += q[H + j]; hn += q[2 * H + j];
    }
    const float hp = hprev ? hprev[idx] : 0.f;
    const float rg = 1.f / (1.f + __expf(-(ir + hr)));
    const float zg = 1.f / (1.f + __expf(-(iz + hz)));
    const float ng = tanhf(in + rg * hn);
    const float hv = (1.f - zg) * ng + zg * hp;
    hout[idx] = hv;
    if (hout_h) hout_h[idx] = (_Float16)hv;
}

// out_h[row] = fp16(relu(concat(a[row], b[row])))   (feeds fc GEMM only)
__global__ void concat_relu(const float* __restrict__ a,
                            const float* __restrict__ b,
                            _Float16* __restrict__ out_h, int rows, int H)
{
    const int idx = blockIdx.x * blockDim.x + threadIdx.x;
    if (idx >= rows * 2 * H) return;
    const int row = idx / (2 * H);
    const int j   = idx - row * 2 * H;
    const float v = (j < H) ? a[(size_t)row * H + j] : b[(size_t)row * H + (j - H)];
    out_h[idx] = (_Float16)fmaxf(v, 0.f);
}

// w[i*4+b] = sigmoid(dot(q[i*4+b,:], k[i,:]))  — one wave per pair
__global__ void attn_w(const float* __restrict__ q, const float* __restrict__ k,
                       float* __restrict__ w, int c, int A)
{
    const int warp = (blockIdx.x * blockDim.x + threadIdx.x) >> 5;
    const int lane = threadIdx.x & 31;
    if (warp >= c * 4) return;
    const float* qr = q + (size_t)warp * A;
    const float* kr = k + (size_t)(warp >> 2) * A;
    float s = 0.f;
    for (int t = lane; t < A; t += 32) s += qr[t] * kr[t];
#pragma unroll
    for (int o = 16; o > 0; o >>= 1) s += __shfl_xor(s, o, 32);
    if (lane == 0) w[warp] = 1.f / (1.f + __expf(-s));
}

// ctx[i,:] = sum_b w[i*4+b] * ch[i*4+b,:]   (+ fp16 mirror for gh GEMM)
__global__ void attn_ctx(const float* __restrict__ w, const float* __restrict__ ch,
                         float* __restrict__ ctx, _Float16* __restrict__ ctx_h,
                         int c, int H)
{
    const int idx = blockIdx.x * blockDim.x + threadIdx.x;
    if (idx >= c * H) return;
    const int i = idx / H;
    const int j = idx - i * H;
    float s = 0.f;
#pragma unroll
    for (int b = 0; b < 4; ++b)
        s += w[i * 4 + b] * ch[(size_t)(i * 4 + b) * H + j];
    ctx[idx] = s;
    ctx_h[idx] = (_Float16)s;
}

// node_out = hid1 @ nc_w2^T + b2 (K=64, N=2); probs = softmax(node_out)[:,1]
__global__ void node_head(const float* __restrict__ hid1,
                          const float* __restrict__ w2, const float* __restrict__ b2,
                          float* __restrict__ node_out, float* __restrict__ probs, int n)
{
    const int i = blockIdx.x * blockDim.x + threadIdx.x;
    if (i >= n) return;
    const float* h = hid1 + (size_t)i * 64;
    float o0 = b2[0], o1 = b2[1];
#pragma unroll
    for (int j = 0; j < 64; ++j) { o0 += w2[j] * h[j]; o1 += w2[64 + j] * h[j]; }
    node_out[2 * i] = o0;
    node_out[2 * i + 1] = o1;
    const float m  = fmaxf(o0, o1);
    const float e0 = __expf(o0 - m), e1 = __expf(o1 - m);
    probs[i] = e1 / (e0 + e1);
}

__global__ void scatter_probs(const float* __restrict__ probs,
                              const float* __restrict__ tid2time,
                              float* __restrict__ second, int n)
{
    const int i = blockIdx.x * blockDim.x + threadIdx.x;
    if (i >= n) return;
    int slot = (int)floorf(tid2time[i] * 0.2f);
    slot = slot < 0 ? 0 : (slot > TTIME - 1 ? TTIME - 1 : slot);
    atomicAdd(&second[slot], probs[i]);
}

// ---------------------------------------------------------------------------
// Batch-1 conv1d(k=3) + 510-step GRU(hidden 256) + linear(512). One block.
// ---------------------------------------------------------------------------
__global__ void cnn_gru(const float* __restrict__ tsr,     // 512
                        const float* __restrict__ second,  // 512
                        const float* __restrict__ conv_w,  // 128x2x3
                        const float* __restrict__ conv_b,  // 128
                        const float* __restrict__ wih,     // 768x128
                        const float* __restrict__ whh,     // 768x256
                        const float* __restrict__ bih,     // 768
                        const float* __restrict__ bhh,     // 768
                        const float* __restrict__ lin_w,   // 512x256
                        const float* __restrict__ lin_b,   // 512
                        float* __restrict__ time_h)        // 512
{
    __shared__ float ts0[TTIME], ts1[TTIME];
    __shared__ float y[THID];
    __shared__ float h[2 * THID];
    __shared__ float gi[6 * THID], gh[6 * THID];
    const int t = threadIdx.x;                    // 256 threads
    for (int i = t; i < TTIME; i += 256) { ts0[i] = tsr[i]; ts1[i] = second[i]; }
    h[t] = 0.f;
    __syncthreads();

    for (int step = 0; step < TTIME - 2; ++step) {
        if (t < THID) {
            const float* w = conv_w + t * 6;
            float s = conv_b[t];
            s += w[0] * ts0[step] + w[1] * ts0[step + 1] + w[2] * ts0[step + 2];
            s += w[3] * ts1[step] + w[4] * ts1[step + 1] + w[5] * ts1[step + 2];
            y[t] = s;
        }
        __syncthreads();
        for (int g = t; g < 6 * THID; g += 256) {
            const float* wi = wih + (size_t)g * THID;
            float s = bih[g];
            for (int j = 0; j < THID; ++j) s += wi[j] * y[j];
            gi[g] = s;
            const float* wh = whh + (size_t)g * (2 * THID);
            float s2 = bhh[g];
            for (int j = 0; j < 2 * THID; ++j) s2 += wh[j] * h[j];
            gh[g] = s2;
        }
        __syncthreads();
        const float rg = 1.f / (1.f + __expf(-(gi[t] + gh[t])));
        const float zg = 1.f / (1.f + __expf(-(gi[256 + t] + gh[256 + t])));
        const float ng = tanhf(gi[512 + t] + rg * gh[512 + t]);
        const float hn = (1.f - zg) * ng + zg * h[t];
        __syncthreads();
        h[t] = hn;
        __syncthreads();
    }

    for (int o = t; o < 4 * THID; o += 256) {
        const float* w = lin_w + (size_t)o * (2 * THID);
        float s = lin_b[o];
        for (int j = 0; j < 2 * THID; ++j) s += w[j] * h[j];
        time_h[o] = s;
    }
}

// final classifier: pred = relu(concat(tree_h,time_h) @ w1^T + b1) @ w2^T + b2
__global__ void classifier(const float* __restrict__ tree_h,  // 512 (h_tree row 0)
                           const float* __restrict__ time_h,  // 512
                           const float* __restrict__ w1,      // 256x1024
                           const float* __restrict__ b1,      // 256
                           const float* __restrict__ w2,      // 4x256
                           const float* __restrict__ b2,      // 4
                           float* __restrict__ pred)          // d_out[0..3]
{
    __shared__ float outv[CLIN];
    __shared__ float hid[256];
    const int t = threadIdx.x;                    // 256 threads
    for (int i = t; i < HID; i += 256) { outv[i] = tree_h[i]; outv[HID + i] = time_h[i]; }
    __syncthreads();
    const float* w = w1 + (size_t)t * CLIN;
    float s = b1[t];
    for (int j = 0; j < CLIN; ++j) s += w[j] * outv[j];
    hid[t] = fmaxf(s, 0.f);
    __syncthreads();
    if (t < 4) {
        const float* wr = w2 + t * 256;
        float p = b2[t];
        for (int j = 0; j < 256; ++j) p += wr[j] * hid[j];
        pred[t] = p;
    }
}

__global__ void labels_to_f32(const int* __restrict__ lab, float* __restrict__ out, int n)
{
    const int i = blockIdx.x * blockDim.x + threadIdx.x;
    if (i < n) out[i] = (float)lab[i];
}

// ---------------------------------------------------------------------------
// Host-side orchestration
// ---------------------------------------------------------------------------
static inline int cdiv(int a, int b) { return (a + b - 1) / b; }

static inline void gemm(const _Float16* A, int lda, const _Float16* W,
                        const float* bias, float* C, _Float16* Ch, int ldc,
                        int M, int Nn, int K, int relu, hipStream_t s)
{
    const int tiles  = ((M + 15) / 16) * (Nn / 64);
    const int blocks = (tiles + 7) / 8;               // 8 waves / block (wave32)
    wmma_gemm<<<blocks, 256, 0, s>>>(A, lda, W, bias, C, Ch, ldc, M, Nn, K, relu);
}

static inline void* carve(char*& p, size_t bytes)
{
    void* r = (void*)p;
    p += (bytes + 255) & ~(size_t)255;                // keep 256B alignment
    return r;
}

extern "C" void kernel_launch(void* const* d_in, const int* in_sizes, int n_in,
                              void* d_out, int out_size, void* d_ws, size_t ws_size,
                              hipStream_t stream)
{
    (void)in_sizes; (void)n_in; (void)out_size; (void)ws_size;

    // ---- inputs (setup_inputs dict order; params in insertion order) ----
    const float* feats     = (const float*)d_in[0];   // (N, 32, 768)
    const float* tsr       = (const float*)d_in[1];   // (1, 512, 1)
    const float* tid2time  = (const float*)d_in[2];   // (N,)
    const int*   labels    = (const int*)  d_in[3];   // (N,)
    int p = 4;
    const float* ne_proj_w = (const float*)d_in[p++];
    const float* ne_proj_b = (const float*)d_in[p++];
    const float* ne_wih_f  = (const float*)d_in[p++];
    const float* ne_whh_f  = (const float*)d_in[p++];
    const float* ne_bih_f  = (const float*)d_in[p++];
    const float* ne_bhh_f  = (const float*)d_in[p++];
    const float* ne_wih_b  = (const float*)d_in[p++];
    const float* ne_whh_b  = (const float*)d_in[p++]; (void)ne_whh_b; // ys_b[0] has h=0
    const float* ne_bih_b  = (const float*)d_in[p++];
    const float* ne_bhh_b  = (const float*)d_in[p++];
    const float* ne_fc_w   = (const float*)d_in[p++];
    const float* ne_fc_b   = (const float*)d_in[p++];
    const float* rv_proj_w = (const float*)d_in[p++];
    const float* rv_proj_b = (const float*)d_in[p++];
    const float* rv_q_w    = (const float*)d_in[p++];
    const float* rv_k_w    = (const float*)d_in[p++];
    const float* rv_wih    = (const float*)d_in[p++];
    const float* rv_whh    = (const float*)d_in[p++];
    const float* rv_bih    = (const float*)d_in[p++];
    const float* rv_bhh    = (const float*)d_in[p++];
    const float* nc_w1     = (const float*)d_in[p++];
    const float* nc_b1     = (const float*)d_in[p++];
    const float* nc_w2     = (const float*)d_in[p++];
    const float* nc_b2     = (const float*)d_in[p++];
    const float* cnn_conv_w= (const float*)d_in[p++];
    const float* cnn_conv_b= (const float*)d_in[p++];
    const float* cnn_wih   = (const float*)d_in[p++];
    const float* cnn_whh   = (const float*)d_in[p++];
    const float* cnn_bih   = (const float*)d_in[p++];
    const float* cnn_bhh   = (const float*)d_in[p++];
    const float* cnn_lin_w = (const float*)d_in[p++];
    const float* cnn_lin_b = (const float*)d_in[p++];
    const float* cl_w1     = (const float*)d_in[p++];
    const float* cl_b1     = (const float*)d_in[p++];
    const float* cl_w2     = (const float*)d_in[p++];
    const float* cl_b2     = (const float*)d_in[p++];

    float* out = (float*)d_out;   // [pred(4) | node_output(2730) | labels(1365)]

    // ---- workspace carve-up (256B-aligned chunks) ----
    char* wsp = (char*)d_ws;
    // fp32 intermediates
    float* h_f     = (float*)carve(wsp, (size_t)NN * HID * 4);
    float* ysb0    = (float*)carve(wsp, (size_t)NN * HID * 4);
    float* gi      = (float*)carve(wsp, (size_t)NN * 3 * HID * 4);
    float* gh      = (float*)carve(wsp, (size_t)NN * 3 * HID * 4);
    float* h_tree  = (float*)carve(wsp, (size_t)NN * HID * 4);
    float* qb      = (float*)carve(wsp, (size_t)1024 * ATTN * 4);
    float* kb      = (float*)carve(wsp, (size_t)256 * ATTN * 4);
    float* wb      = (float*)carve(wsp, (size_t)1024 * 4);
    float* ctxb    = (float*)carve(wsp, (size_t)256 * HID * 4);
    float* hid1    = (float*)carve(wsp, (size_t)NN * 64 * 4);
    float* probs   = (float*)carve(wsp, (size_t)NN * 4);
    float* second  = (float*)carve(wsp, (size_t)TTIME * 4);
    float* time_h  = (float*)carve(wsp, (size_t)TTIME * 4);
    // fp16 activation staging
    _Float16* feats_h   = (_Float16*)carve(wsp, (size_t)NN * TSEQ * FEAT * 2);
    _Float16* xbuf_h    = (_Float16*)carve(wsp, (size_t)NN * HID * 2);
    _Float16* h_f_h     = (_Float16*)carve(wsp, (size_t)NN * HID * 2);
    _Float16* outcat_h  = (_Float16*)carve(wsp, (size_t)NN * 2 * HID * 2);
    _Float16* emb_h     = (_Float16*)carve(wsp, (size_t)NN * HID * 2);
    _Float16* proj_rv_h = (_Float16*)carve(wsp, (size_t)NN * HID * 2);
    _Float16* h_tree_h  = (_Float16*)carve(wsp, (size_t)NN * HID * 2);
    _Float16* ctxb_h    = (_Float16*)carve(wsp, (size_t)256 * HID * 2);
    // fp16 weight staging
    _Float16* w_proj_h  = (_Float16*)carve(wsp, (size_t)HID * FEAT * 2);
    _Float16* w_ihf_h   = (_Float16*)carve(wsp, (size_t)3 * HID * HID * 2);
    _Float16* w_hhf_h   = (_Float16*)carve(wsp, (size_t)3 * HID * HID * 2);
    _Float16* w_ihb_h   = (_Float16*)carve(wsp, (size_t)3 * HID * HID * 2);
    _Float16* w_fc_h    = (_Float16*)carve(wsp, (size_t)HID * 2 * HID * 2);
    _Float16* w_rvp_h   = (_Float16*)carve(wsp, (size_t)HID * HID * 2);
    _Float16* w_rvq_h   = (_Float16*)carve(wsp, (size_t)ATTN * HID * 2);
    _Float16* w_rvk_h   = (_Float16*)carve(wsp, (size_t)ATTN * HID * 2);
    _Float16* w_rvih_h  = (_Float16*)carve(wsp, (size_t)3 * HID * HID * 2);
    _Float16* w_rvhh_h  = (_Float16*)carve(wsp, (size_t)3 * HID * HID * 2);
    _Float16* w_nc1_h   = (_Float16*)carve(wsp, (size_t)64 * HID * 2);

    const int TPB = 256;
#define CVT(src, dst, n) f32_to_f16<<<cdiv((n), TPB), TPB, 0, stream>>>((src), (dst), (n))

    // ---- one-time fp16 staging (weights reused up to 32x) ----
    CVT(feats,     feats_h,  NN * TSEQ * FEAT);
    CVT(ne_proj_w, w_proj_h, HID * FEAT);
    CVT(ne_wih_f,  w_ihf_h,  3 * HID * HID);
    CVT(ne_whh_f,  w_hhf_h,  3 * HID * HID);
    CVT(ne_wih_b,  w_ihb_h,  3 * HID * HID);
    CVT(ne_fc_w,   w_fc_h,   HID * 2 * HID);
    CVT(rv_proj_w, w_rvp_h,  HID * HID);
    CVT(rv_q_w,    w_rvq_h,  ATTN * HID);
    CVT(rv_k_w,    w_rvk_h,  ATTN * HID);
    CVT(rv_wih,    w_rvih_h, 3 * HID * HID);
    CVT(rv_whh,    w_rvhh_h, 3 * HID * HID);
    CVT(nc_w1,     w_nc1_h,  64 * HID);
#undef CVT

    // ================= node embedding (forward GRU over 32 steps) ==========
    zero_f32<<<cdiv(NN * HID, TPB), TPB, 0, stream>>>(h_f, NN * HID);
    zero_f16<<<cdiv(NN * HID, TPB), TPB, 0, stream>>>(h_f_h, NN * HID);
    for (int t = 0; t < TSEQ; ++t) {
        // x_t = relu(feats[:,t,:] @ ne_proj_w^T + b)   (1365x512, K=768) -> fp16
        gemm(feats_h + (size_t)t * FEAT, TSEQ * FEAT, w_proj_h, ne_proj_b,
             nullptr, xbuf_h, HID, NN, HID, FEAT, 1, stream);
        // gi = x_t @ wih_f^T ; gh = h @ whh_f^T        (1365x1536, K=512)
        gemm(xbuf_h, HID, w_ihf_h, nullptr, gi, nullptr, 3 * HID, NN, 3 * HID, HID, 0, stream);
        gemm(h_f_h,  HID, w_hhf_h, nullptr, gh, nullptr, 3 * HID, NN, 3 * HID, HID, 0, stream);
        gru_gate<<<cdiv(NN * HID, TPB), TPB, 0, stream>>>(
            gi, gh, ne_bih_f, ne_bhh_f, h_f, h_f, h_f_h, NN, HID);
    }
    // ys_b[0] == gru_cell(x[:,31,:], h=0) with backward params (scan collapse)
    gemm(feats_h + (size_t)(TSEQ - 1) * FEAT, TSEQ * FEAT, w_proj_h, ne_proj_b,
         nullptr, xbuf_h, HID, NN, HID, FEAT, 1, stream);
    gemm(xbuf_h, HID, w_ihb_h, nullptr, gi, nullptr, 3 * HID, NN, 3 * HID, HID, 0, stream);
    gru_gate<<<cdiv(NN * HID, TPB), TPB, 0, stream>>>(
        gi, nullptr, ne_bih_b, ne_bhh_b, nullptr, ysb0, nullptr, NN, HID);

    // emb = relu(concat(h_f, ysb0)) @ fc^T + fc_b      (1365x512, K=1024)
    concat_relu<<<cdiv(NN * 2 * HID, TPB), TPB, 0, stream>>>(h_f, ysb0, outcat_h, NN, HID);
    gemm(outcat_h, 2 * HID, w_fc_h, ne_fc_b, nullptr, emb_h, HID, NN, HID, 2 * HID, 0, stream);

    // ================= RvNN tree ===========================================
    gemm(emb_h, HID, w_rvp_h, rv_proj_b, nullptr, proj_rv_h, HID, NN, HID, HID, 0, stream);

    // bottom level (s=341, c=1024): h = gru_cell(proj, h=0)
    gemm(proj_rv_h + (size_t)341 * HID, HID, w_rvih_h, nullptr,
         gi, nullptr, 3 * HID, 1024, 3 * HID, HID, 0, stream);
    gru_gate<<<cdiv(1024 * HID, TPB), TPB, 0, stream>>>(
        gi, nullptr, rv_bih, rv_bhh, nullptr,
        h_tree + (size_t)341 * HID, h_tree_h + (size_t)341 * HID, 1024, HID);

    const int lstart[5] = {85, 21, 5, 1, 0};
    const int lsize[5]  = {256, 64, 16, 4, 1};
    for (int l = 0; l < 5; ++l) {
        const int s = lstart[l], c = lsize[l];
        const _Float16* pl_h = proj_rv_h + (size_t)s * HID;
        const float*    chf  = h_tree   + (size_t)(4 * s + 1) * HID;  // contiguous kids
        const _Float16* chh  = h_tree_h + (size_t)(4 * s + 1) * HID;
        // attention
        gemm(chh,  HID, w_rvq_h, nullptr, qb, nullptr, ATTN, 4 * c, ATTN, HID, 0, stream);
        gemm(pl_h, HID, w_rvk_h, nullptr, kb, nullptr, ATTN, c,     ATTN, HID, 0, stream);
        attn_w<<<cdiv(4 * c * 32, TPB), TPB, 0, stream>>>(qb, kb, wb, c, ATTN);
        attn_ctx<<<cdiv(c * HID, TPB), TPB, 0, stream>>>(wb, chf, ctxb, ctxb_h, c, HID);
        // h[s:s+c] = gru_cell(x=pl, h=ctx)
        gemm(pl_h,   HID, w_rvih_h, nullptr, gi, nullptr, 3 * HID, c, 3 * HID, HID, 0, stream);
        gemm(ctxb_h, HID, w_rvhh_h, nullptr, gh, nullptr, 3 * HID, c, 3 * HID, HID, 0, stream);
        gru_gate<<<cdiv(c * HID, TPB), TPB, 0, stream>>>(
            gi, gh, rv_bih, rv_bhh, ctxb,
            h_tree + (size_t)s * HID, h_tree_h + (size_t)s * HID, c, HID);
    }

    // ================= node classifier + scatter ===========================
    gemm(emb_h, HID, w_nc1_h, nc_b1, hid1, nullptr, 64, NN, 64, HID, 1, stream);
    node_head<<<cdiv(NN, TPB), TPB, 0, stream>>>(hid1, nc_w2, nc_b2,
                                                 out + 4, probs, NN);
    zero_f32<<<cdiv(TTIME, TPB), TPB, 0, stream>>>(second, TTIME);
    scatter_probs<<<cdiv(NN, TPB), TPB, 0, stream>>>(probs, tid2time, second, NN);

    // ================= timeseries conv + GRU (batch 1, sequential) =========
    cnn_gru<<<1, 256, 0, stream>>>(tsr, second, cnn_conv_w, cnn_conv_b,
                                   cnn_wih, cnn_whh, cnn_bih, cnn_bhh,
                                   cnn_lin_w, cnn_lin_b, time_h);

    // ================= final classifier + label passthrough ================
    classifier<<<1, 256, 0, stream>>>(h_tree /*row 0*/, time_h,
                                      cl_w1, cl_b1, cl_w2, cl_b2, out);
    labels_to_f32<<<cdiv(NN, TPB), TPB, 0, stream>>>(labels, out + 4 + 2 * NN, NN);
}
